// InformerEncoderLayer_17205638988530
// MI455X (gfx1250) — matrix operand order repaired and checked by reading, not compiled, not run on hardware
//
#include <hip/hip_runtime.h>

// ---------------- types / helpers ----------------
typedef __attribute__((ext_vector_type(16))) __bf16 v16bf;
typedef __attribute__((ext_vector_type(8)))  float  v8f;

#define D_MODEL 512
#define D_FF    2048
#define SEQ_L   2048
#define BATCH   4
#define NROWS   (BATCH * SEQ_L)   // 8192
#define HEADS   8
#define HDIM    64
#define TOPU    8

__device__ __forceinline__ __bf16 f2bf(float f) {
    unsigned u = __float_as_uint(f);
    u += 0x7FFFu + ((u >> 16) & 1u);
    unsigned short s = (unsigned short)(u >> 16);
    return __builtin_bit_cast(__bf16, s);
}
__device__ __forceinline__ float bf2f(__bf16 b) {
    unsigned short s = __builtin_bit_cast(unsigned short, b);
    return __uint_as_float(((unsigned)s) << 16);
}

// ---------------- pack kernels ----------------
__global__ void cvt_bf16_kernel(const float* __restrict__ in, __bf16* __restrict__ out, int n) {
    int i = blockIdx.x * 256 + threadIdx.x;
    if (i < n) out[i] = f2bf(in[i]);
}

// W [K][N] (f32)  ->  Wt [N][K] (bf16)   (transposed pack: WMMA B columns become contiguous)
__global__ void cvt_bf16_T_kernel(const float* __restrict__ in, __bf16* __restrict__ out,
                                  int K, int N) {
    int i = blockIdx.x * 256 + threadIdx.x;
    if (i < K * N) {
        int n = i / K, k = i % K;       // i indexes output [n][k]
        out[i] = f2bf(in[(size_t)k * N + n]);
    }
}

// Wc [co][ci][3] (f32)  ->  Wcb [t][co][ci] (bf16)  (transposed per tap)
__global__ void cvt_wc_kernel(const float* __restrict__ Wc, __bf16* __restrict__ out) {
    int i = blockIdx.x * 256 + threadIdx.x;
    if (i < 3 * D_MODEL * D_MODEL) {
        int t = i / (D_MODEL * D_MODEL);
        int r = i % (D_MODEL * D_MODEL);
        int co = r / D_MODEL, ci = r % D_MODEL;   // output [t][co][ci]
        out[i] = f2bf(Wc[(co * D_MODEL + ci) * 3 + t]);
    }
}

// ---------------- ProbSparse attention ----------------
// grid: 32 (B*H) * 128 (query tiles of 16) blocks, 1 wave each.
__global__ __launch_bounds__(32) void attn_topk_kernel(const __bf16* __restrict__ xb,
                                                       __bf16* __restrict__ attn_b) {
    __shared__ float sc[16][257];
    const int lane = threadIdx.x;
    const int half = lane >> 4;
    const int col  = lane & 15;
    const int qt = blockIdx.x & 127;       // query tile
    const int n  = blockIdx.x >> 7;        // B*H index
    const int h  = n & 7;
    const int bb = n >> 3;
    const size_t base = (size_t)bb * (SEQ_L * D_MODEL) + (size_t)h * HDIM;

    // A fragments: q rows qt*16..+15, K = head dims 0..63 (2 frags of 32)
    const int mrow = qt * 16 + col;
    const __bf16* qp = xb + base + (size_t)mrow * D_MODEL;
    v16bf a0, a1;
    #pragma unroll
    for (int i = 0; i < 8; i++) {
        a0[i]     = qp[half * 8 + i];
        a0[8 + i] = qp[16 + half * 8 + i];
        a1[i]     = qp[32 + half * 8 + i];
        a1[8 + i] = qp[48 + half * 8 + i];
    }

    float tv[TOPU]; int ti8[TOPU];
    #pragma unroll
    for (int j = 0; j < TOPU; j++) { tv[j] = -3.0e38f; ti8[j] = 0; }

    const float scaling = 0.04419417382415922f;  // 1/sqrt(512)

    for (int c0 = 0; c0 < SEQ_L; c0 += 256) {
        for (int nt = 0; nt < 16; nt++) {
            const int krow = c0 + nt * 16 + col;     // B column = key row
            const __bf16* kp = xb + base + (size_t)krow * D_MODEL;
            v16bf b0, b1;
            #pragma unroll
            for (int i = 0; i < 16; i++) {
                b0[i] = kp[half * 8 + (i < 8 ? i : i + 8)];
                b1[i] = kp[32 + half * 8 + (i < 8 ? i : i + 8)];
            }
            v8f acc = {};
            acc = __builtin_amdgcn_wmma_f32_16x16x32_bf16(false, a0, false, b0, (short)0, acc, false, false);
            acc = __builtin_amdgcn_wmma_f32_16x16x32_bf16(false, a1, false, b1, (short)0, acc, false, false);
            #pragma unroll
            for (int r = 0; r < 8; r++)
                sc[r + 8 * half][nt * 16 + col] = acc[r] * scaling;
        }
        __syncthreads();
        if (lane < 16) {
            #pragma unroll 4
            for (int j = 0; j < 256; j++) {
                float v = sc[lane][j];
                if (v > tv[TOPU - 1]) {
                    float nv = v; int ni = c0 + j;
                    #pragma unroll
                    for (int p = 0; p < TOPU; p++) {
                        if (nv > tv[p]) {
                            float t0 = tv[p]; int t1 = ti8[p];
                            tv[p] = nv; ti8[p] = ni; nv = t0; ni = t1;
                        }
                    }
                }
            }
        }
        __syncthreads();
    }

    // softmax over top-8 (tv is sorted descending -> tv[0] is max)
    float pv[TOPU];
    #pragma unroll
    for (int j = 0; j < TOPU; j++) pv[j] = 0.f;
    if (lane < 16) {
        float mx = tv[0], s = 0.f;
        #pragma unroll
        for (int j = 0; j < TOPU; j++) { pv[j] = expf(tv[j] - mx); s += pv[j]; }
        float inv = 1.f / s;
        #pragma unroll
        for (int j = 0; j < TOPU; j++) pv[j] *= inv;
    }

    // gather value rows, weighted sum; each lane writes 2 head-dims
    const int j0 = lane * 2;
    for (int r = 0; r < 16; r++) {
        float o0 = 0.f, o1 = 0.f;
        #pragma unroll
        for (int j = 0; j < TOPU; j++) {
            float p  = __shfl(pv[j], r, 32);
            int   vi = __shfl(ti8[j], r, 32);
            const __bf16* vp = xb + base + (size_t)vi * D_MODEL + j0;
            o0 += p * bf2f(vp[0]);
            o1 += p * bf2f(vp[1]);
        }
        __bf16* op = attn_b + (size_t)bb * (SEQ_L * D_MODEL)
                   + (size_t)(qt * 16 + r) * D_MODEL + h * HDIM + j0;
        op[0] = f2bf(o0);
        op[1] = f2bf(o1);
    }
}

// ---------------- generic WMMA GEMM (64x16 per wave; B transposed [N][K]) ----------------
// C[m][n] = sum_k A[m][k]*Bt[n][k] + bias[n] (+ res[m][n])
// mode 0: store f32   mode 1: exact GELU then store bf16
__global__ __launch_bounds__(256) void gemm_bf16_kernel(const __bf16* __restrict__ A,
                                                        const __bf16* __restrict__ Bt,
                                                        const float* __restrict__ bias,
                                                        const float* __restrict__ res,
                                                        float* __restrict__ outF,
                                                        __bf16* __restrict__ outB,
                                                        int M, int N, int K, int mode) {
    const int lane = threadIdx.x;
    const int t = blockIdx.x * 8 + threadIdx.y;    // wave-tile id (64 rows x 16 cols)
    const int tilesN = N >> 4;
    const int tilesM = M >> 6;
    if (t >= tilesM * tilesN) return;
    const int m0 = (t / tilesN) << 6;
    const int n0 = (t % tilesN) << 4;
    const int half = lane >> 4, col = lane & 15;

    const __bf16* ap0 = A + (size_t)(m0 + col) * K;
    const __bf16* bp0 = Bt + (size_t)(n0 + col) * K;   // column n is a contiguous row of Bt
    const size_t astride = (size_t)16 * K;

    v8f accs[4] = {v8f{}, v8f{}, v8f{}, v8f{}};
    for (int k0 = 0; k0 < K; k0 += 32) {
        // B fragment: 2 contiguous 16B runs, reused by 4 WMMAs
        const __bf16* bp = bp0 + k0 + half * 8;
        v16bf bm;
        #pragma unroll
        for (int i = 0; i < 16; i++) bm[i] = bp[i < 8 ? i : i + 8];
        if (k0 + 32 < K) __builtin_prefetch(bp0 + k0 + 64, 0, 3);
        #pragma unroll
        for (int s = 0; s < 4; s++) {
            const __bf16* ap = ap0 + (size_t)s * astride + k0;
            v16bf a;
            #pragma unroll
            for (int i = 0; i < 8; i++) {
                a[i]     = ap[half * 8 + i];
                a[8 + i] = ap[16 + half * 8 + i];
            }
            accs[s] = __builtin_amdgcn_wmma_f32_16x16x32_bf16(false, a, false, bm, (short)0, accs[s], false, false);
        }
    }

    const int nn = n0 + col;
    const float bv = bias[nn];
    #pragma unroll
    for (int s = 0; s < 4; s++) {
        #pragma unroll
        for (int r = 0; r < 8; r++) {
            const int m = m0 + s * 16 + r + 8 * half;
            const size_t o = (size_t)m * N + nn;
            float v = accs[s][r] + bv;
            if (res) v += res[o];
            if (mode == 1) {
                float g = 0.5f * v * (1.0f + erff(v * 0.70710678118654752f));
                outB[o] = f2bf(g);
            } else {
                outF[o] = v;
            }
        }
    }
}

// ---------------- LayerNorm ----------------
__global__ __launch_bounds__(128) void layernorm_kernel(const float* __restrict__ in,
                                                        const float* __restrict__ g,
                                                        const float* __restrict__ b,
                                                        float* __restrict__ outF,
                                                        __bf16* __restrict__ outB) {
    __shared__ float red[128];
    const int row = blockIdx.x, tid = threadIdx.x;
    const float* p = in + (size_t)row * D_MODEL;

    float s = 0.f;
    for (int i = tid; i < D_MODEL; i += 128) s += p[i];
    red[tid] = s; __syncthreads();
    for (int off = 64; off > 0; off >>= 1) { if (tid < off) red[tid] += red[tid + off]; __syncthreads(); }
    const float mean = red[0] * (1.0f / D_MODEL);
    __syncthreads();

    float v = 0.f;
    for (int i = tid; i < D_MODEL; i += 128) { float d = p[i] - mean; v += d * d; }
    red[tid] = v; __syncthreads();
    for (int off = 64; off > 0; off >>= 1) { if (tid < off) red[tid] += red[tid + off]; __syncthreads(); }
    const float rstd = rsqrtf(red[0] * (1.0f / D_MODEL) + 1e-5f);

    for (int i = tid; i < D_MODEL; i += 128) {
        float o = (p[i] - mean) * rstd * g[i] + b[i];
        if (outF) outF[(size_t)row * D_MODEL + i] = o;
        if (outB) outB[(size_t)row * D_MODEL + i] = f2bf(o);
    }
}

// ---------------- Conv1d(k=3,p=1) + ELU + maxpool2 (fused, 64x16 per wave) ----------------
// X: [B*L, 512] bf16 (LN2 output); Wcb: [t][co][ci] bf16; out: [B,1024,512] f32
__global__ __launch_bounds__(256) void conv_elu_pool_kernel(const __bf16* __restrict__ X,
                                                            const __bf16* __restrict__ Wcb,
                                                            const float* __restrict__ bc,
                                                            float* __restrict__ out) {
    const int lane = threadIdx.x;
    const int t = blockIdx.x * 8 + threadIdx.y;
    const int tilesN = D_MODEL >> 4;  // 32
    const int tilesM = NROWS >> 6;    // 128
    if (t >= tilesM * tilesN) return;
    const int m0 = (t / tilesN) << 6;
    const int n0 = (t % tilesN) << 4;
    const int half = lane >> 4, col = lane & 15;
    const __bf16 z = __builtin_bit_cast(__bf16, (unsigned short)0);

    v8f accs[4] = {v8f{}, v8f{}, v8f{}, v8f{}};
    #pragma unroll
    for (int tt = 0; tt < 3; tt++) {
        const int sh = tt - 1;
        // weight column n0+col for this tap: contiguous in ci
        const __bf16* bp0 = Wcb + (size_t)tt * D_MODEL * D_MODEL + (size_t)(n0 + col) * D_MODEL;
        for (int k0 = 0; k0 < D_MODEL; k0 += 32) {
            const __bf16* bp = bp0 + k0 + half * 8;
            v16bf bm;
            #pragma unroll
            for (int i = 0; i < 16; i++) bm[i] = bp[i < 8 ? i : i + 8];
            #pragma unroll
            for (int s = 0; s < 4; s++) {
                const int m = m0 + s * 16 + col;
                const int ls = (m & (SEQ_L - 1)) + sh;
                const bool ok = (ls >= 0) && (ls < SEQ_L);
                const __bf16* ap = X + (size_t)(ok ? (m + sh) : m) * D_MODEL + k0;  // clamped-safe
                v16bf a;
                #pragma unroll
                for (int i = 0; i < 8; i++) {
                    __bf16 x0 = ap[half * 8 + i];
                    __bf16 x1 = ap[16 + half * 8 + i];
                    a[i]     = ok ? x0 : z;
                    a[8 + i] = ok ? x1 : z;
                }
                accs[s] = __builtin_amdgcn_wmma_f32_16x16x32_bf16(false, a, false, bm, (short)0, accs[s], false, false);
            }
        }
    }

    // fused ELU + maxpool2: D rows M=2r,2r+1 are acc[2r],acc[2r+1]
    const int nn = n0 + col;
    const float bv = bc[nn];
    #pragma unroll
    for (int s = 0; s < 4; s++) {
        const int ms = m0 + s * 16;
        const int bidx = ms >> 11;            // / 2048
        const int l0 = ms & (SEQ_L - 1);
        #pragma unroll
        for (int r = 0; r < 4; r++) {
            float v = fmaxf(accs[s][2 * r], accs[s][2 * r + 1]) + bv;
            float e = (v > 0.f) ? v : expm1f(v);
            const int l2 = (l0 >> 1) + half * 4 + r;
            out[(size_t)bidx * (SEQ_L / 2) * D_MODEL + (size_t)l2 * D_MODEL + nn] = e;
        }
    }
}

// ---------------- launcher ----------------
extern "C" void kernel_launch(void* const* d_in, const int* in_sizes, int n_in,
                              void* d_out, int out_size, void* d_ws, size_t ws_size,
                              hipStream_t stream) {
    const float* x     = (const float*)d_in[0];
    const float* Wo    = (const float*)d_in[1];
    const float* bo    = (const float*)d_in[2];
    const float* ln1_g = (const float*)d_in[3];
    const float* ln1_b = (const float*)d_in[4];
    const float* ln2_g = (const float*)d_in[5];
    const float* ln2_b = (const float*)d_in[6];
    const float* W1    = (const float*)d_in[7];
    const float* b1    = (const float*)d_in[8];
    const float* W2    = (const float*)d_in[9];
    const float* b2    = (const float*)d_in[10];
    const float* Wc    = (const float*)d_in[11];
    const float* bc    = (const float*)d_in[12];
    float* out = (float*)d_out;

    char* ws = (char*)d_ws;
    size_t off = 0;
    auto alloc = [&](size_t bytes) { char* p = ws + off; off += (bytes + 255) & ~(size_t)255; return p; };

    __bf16* xb     = (__bf16*)alloc((size_t)NROWS * D_MODEL * 2);
    __bf16* attn_b = (__bf16*)alloc((size_t)NROWS * D_MODEL * 2);
    float*  pre1   = (float*) alloc((size_t)NROWS * D_MODEL * 4);
    float*  x1     = (float*) alloc((size_t)NROWS * D_MODEL * 4);
    __bf16* x1b    = (__bf16*)alloc((size_t)NROWS * D_MODEL * 2);
    __bf16* hb     = (__bf16*)alloc((size_t)NROWS * D_FF * 2);
    float*  pre2   = (float*) alloc((size_t)NROWS * D_MODEL * 4);
    __bf16* x2b    = (__bf16*)alloc((size_t)NROWS * D_MODEL * 2);
    __bf16* WoT    = (__bf16*)alloc((size_t)D_MODEL * D_MODEL * 2);   // [N][K]
    __bf16* W1T    = (__bf16*)alloc((size_t)D_MODEL * D_FF * 2);      // [N=2048][K=512]
    __bf16* W2T    = (__bf16*)alloc((size_t)D_FF * D_MODEL * 2);      // [N=512][K=2048]
    __bf16* Wcb    = (__bf16*)alloc((size_t)3 * D_MODEL * D_MODEL * 2); // [t][co][ci]
    (void)ws_size;

    // 1. pack to bf16 (weights transposed so WMMA B-columns are contiguous)
    {
        int n;
        n = NROWS * D_MODEL;   cvt_bf16_kernel<<<(n + 255) / 256, 256, 0, stream>>>(x, xb, n);
        n = D_MODEL * D_MODEL; cvt_bf16_T_kernel<<<(n + 255) / 256, 256, 0, stream>>>(Wo, WoT, D_MODEL, D_MODEL);
        n = D_MODEL * D_FF;    cvt_bf16_T_kernel<<<(n + 255) / 256, 256, 0, stream>>>(W1, W1T, D_MODEL, D_FF);
        n = D_FF * D_MODEL;    cvt_bf16_T_kernel<<<(n + 255) / 256, 256, 0, stream>>>(W2, W2T, D_FF, D_MODEL);
        n = 3 * D_MODEL * D_MODEL; cvt_wc_kernel<<<(n + 255) / 256, 256, 0, stream>>>(Wc, Wcb);
    }

    // 2. ProbSparse attention -> attn_b (bf16)
    attn_topk_kernel<<<BATCH * HEADS * (SEQ_L / 16), 32, 0, stream>>>(xb, attn_b);

    // 3. out_proj + residual:  pre1 = x + attn_b @ Wo + bo
    {
        int tiles = (NROWS / 64) * (D_MODEL / 16);
        gemm_bf16_kernel<<<(tiles + 7) / 8, dim3(32, 8), 0, stream>>>(
            attn_b, WoT, bo, x, pre1, nullptr, NROWS, D_MODEL, D_MODEL, 0);
    }

    // 4. LN1 -> x1 (f32) + x1b (bf16)
    layernorm_kernel<<<NROWS, 128, 0, stream>>>(pre1, ln1_g, ln1_b, x1, x1b);

    // 5. FFN up + GELU -> hb (bf16)
    {
        int tiles = (NROWS / 64) * (D_FF / 16);
        gemm_bf16_kernel<<<(tiles + 7) / 8, dim3(32, 8), 0, stream>>>(
            x1b, W1T, b1, nullptr, nullptr, hb, NROWS, D_FF, D_MODEL, 1);
    }

    // 6. FFN down + residual: pre2 = x1 + hb @ W2 + b2
    {
        int tiles = (NROWS / 64) * (D_MODEL / 16);
        gemm_bf16_kernel<<<(tiles + 7) / 8, dim3(32, 8), 0, stream>>>(
            hb, W2T, b2, x1, pre2, nullptr, NROWS, D_MODEL, D_FF, 0);
    }

    // 7. LN2 -> x2b (bf16 only)
    layernorm_kernel<<<NROWS, 128, 0, stream>>>(pre2, ln2_g, ln2_b, nullptr, x2b);

    // 8. conv1d + ELU + maxpool2 -> d_out
    {
        int tiles = (NROWS / 64) * (D_MODEL / 16);
        conv_elu_pool_kernel<<<(tiles + 7) / 8, dim3(32, 8), 0, stream>>>(x2b, Wcb, bc, out);
    }
    (void)in_sizes; (void)n_in; (void)out_size;
}
